// GraphLSTMBSL_22033182228480
// MI455X (gfx1250) — matrix operand order repaired and checked by reading, not compiled, not run on hardware
//
#include <hip/hip_runtime.h>

// ---------------------------------------------------------------------------
// GraphLSTM on MI455X (gfx1250, wave32, WMMA).
// Gate GEMMs use v_wmma_f32_16x16x32_f16 (f16 operands, f32 accumulate).
// Carry states (h, c) kept in f32; f16 shadow copies feed the WMMA A operands.
// Per timestep: graph_step -> reduce_gh -> cell_step (stream-ordered).
// ---------------------------------------------------------------------------

typedef _Float16 h16;
typedef __attribute__((ext_vector_type(16))) _Float16 v16h;
typedef __attribute__((ext_vector_type(8)))  _Float16 v8h;
typedef __attribute__((ext_vector_type(8)))  float    v8f;

#define SEQ   32
#define NN    65536
#define GH    64
#define CH    64
#define OUTW  5
#define WPB   8            // waves per block (256 threads)
#define NODES_PER_BLOCK 128
#define NBLK  (NN / NODES_PER_BLOCK)   // 512

// ---- WMMA helper -----------------------------------------------------------
__device__ __forceinline__ v8f wmma16(v16h a, v16h b, v8f c) {
  // D = A(16x32 f16) * B(32x16 f16) + C(16x16 f32)
  return __builtin_amdgcn_wmma_f32_16x16x32_f16(
      /*neg_a=*/false, a, /*neg_b=*/false, b,
      /*c_mod=*/(short)0, c, /*reuse_a=*/false, /*reuse_b=*/false);
}

// A-operand (16-bit, 16x32) layout: lane L holds row m=L&15;
// halves e=0..7 -> K = kbase+e, e=8..15 -> K = kbase+8+e-? => kbase+16..kbase+23,
// with kbase = (L>>4)*8.  Load = two contiguous 16B chunks at kbase, kbase+16.
__device__ __forceinline__ v16h loadA16(const h16* row, int kbase, int koff) {
  v8h lo = *(const v8h*)(row + koff + kbase);
  v8h hi = *(const v8h*)(row + koff + kbase + 16);
  v16h r;
#pragma unroll
  for (int i = 0; i < 8; ++i) { r[i] = lo[i]; r[8 + i] = hi[i]; }
  return r;
}

// B operand pre-packed so each lane's 16 halves are contiguous (32B per lane).
__device__ __forceinline__ v16h loadB16(const h16* pk, int blk, int lane) {
  const v8h* p = (const v8h*)(pk + (((size_t)blk * 32 + lane) << 4));
  v8h lo = p[0], hi = p[1];
  v16h r;
#pragma unroll
  for (int i = 0; i < 8; ++i) { r[i] = lo[i]; r[8 + i] = hi[i]; }
  return r;
}

__device__ __forceinline__ float sigmoidf_(float x) {
  return 1.0f / (1.0f + __expf(-x));
}

// ---- init: copy h0/c0 states into ws (f32 + f16 shadows) -------------------
__global__ void init_states(const float* __restrict__ ch0, const float* __restrict__ cc0,
                            const float* __restrict__ gh0, const float* __restrict__ gc0,
                            float* ghf, float* gcf, float* chf, float* ccf,
                            h16* gh16, h16* ch16, int n) {
  int i = blockIdx.x * blockDim.x + threadIdx.x;
  if (i < n) {
    float gh = gh0[i], ch = ch0[i];
    ghf[i] = gh; gcf[i] = gc0[i]; chf[i] = ch; ccf[i] = cc0[i];
    gh16[i] = (h16)gh; ch16[i] = (h16)ch;
  }
}

// ---- weight packing into the WMMA B-operand register layout ----------------
// B[k][n] for a tile: value at (lane, e) = Bmat[kc*32 + (lane>>4)*8 + e + (e>=8 ? 8 : 0)][ct*16 + (lane&15)]
// For gates, Bmat = W^T (W is [256, K] row-major): Bmat[k][n] = W[n*K + k].
__device__ __forceinline__ void pack_wT(const float* W, int K, int CT, h16* dst, int tid) {
  int total = (K / 32) * CT * 512;
  for (int idx = tid; idx < total; idx += 256) {
    int e = idx & 15, lane = (idx >> 4) & 31, blk = idx >> 9;
    int ct = blk % CT, kc = blk / CT;
    int k = kc * 32 + ((lane >> 4) << 3) + e + (e >= 8 ? 8 : 0);
    int n = ct * 16 + (lane & 15);
    dst[idx] = (h16)W[n * K + k];
  }
}

__global__ void pack_weights(const float* __restrict__ WihG, const float* __restrict__ WhhG,
                             const float* __restrict__ bihG, const float* __restrict__ bhhG,
                             const float* __restrict__ wgraph,
                             const float* __restrict__ WihC, const float* __restrict__ WhhC,
                             const float* __restrict__ bihC, const float* __restrict__ bhhC,
                             const float* __restrict__ Wout,
                             h16* pkIHg, h16* pkHHg, h16* pkIHc, h16* pkHHc,
                             h16* pkWg, h16* pkOut, float* bg, float* bc) {
  int tid = threadIdx.x;
  pack_wT(WihG, 32, 16, pkIHg, tid);   // graph Wih^T: K=32, 16 col tiles
  pack_wT(WhhG, 64, 16, pkHHg, tid);   // graph Whh^T: K=64
  pack_wT(WihC, 64, 16, pkIHc, tid);   // cell Wih^T
  pack_wT(WhhC, 64, 16, pkHHc, tid);   // cell Whh^T
  // w_graph used directly as B ([64,32] K-major): Bmat[k][n] = wgraph[k*32+n]
  for (int idx = tid; idx < 2 * 2 * 512; idx += 256) {
    int e = idx & 15, lane = (idx >> 4) & 31, blk = idx >> 9;
    int ct = blk % 2, kc = blk / 2;
    int k = kc * 32 + ((lane >> 4) << 3) + e + (e >= 8 ? 8 : 0);
    int n = ct * 16 + (lane & 15);
    pkWg[idx] = (h16)wgraph[k * 32 + n];
  }
  // W_out^T padded to 16 columns (only 5 valid): Bmat[k][n] = Wout[n*64+k]
  for (int idx = tid; idx < 2 * 512; idx += 256) {
    int e = idx & 15, lane = (idx >> 4) & 31, kc = idx >> 9;
    int k = kc * 32 + ((lane >> 4) << 3) + e + (e >= 8 ? 8 : 0);
    int n = lane & 15;
    pkOut[idx] = (n < OUTW) ? (h16)Wout[n * 64 + k] : (h16)0.0f;
  }
  bg[tid] = bihG[tid] + bhhG[tid];
  bc[tid] = bihC[tid] + bhhC[tid];
}

// ---- per-step kernel 1: graph LSTM cell + deterministic partial gh sums ----
__global__ void __launch_bounds__(256) graph_step(
    const float* __restrict__ in_data, int t,
    const float* __restrict__ Wgemb, const float* __restrict__ bgemb,
    h16* gh16, float* ghf, float* gcf,
    const h16* __restrict__ pkIHg, const h16* __restrict__ pkHHg,
    const float* __restrict__ bg, float* __restrict__ partials) {
  __shared__ float sWg[64], sBg[32];
  __shared__ float sSum[WPB * GH];

  int tid = threadIdx.x, lane = tid & 31, widx = tid >> 5;
  if (tid < 64) sWg[tid] = Wgemb[tid];
  if (tid < 32) sBg[tid] = bgemb[tid];
  __syncthreads();

  int base = (blockIdx.x * WPB + widx) * 16;
  int mrow = lane & 15, hs = lane >> 4, kbase = hs * 8;
  int node = base + mrow;

  // gemb = relu(frame @ W_gemb^T + b), generated directly in A layout
  float f0 = in_data[((size_t)t * NN + node) * 2 + 0];
  float f1 = in_data[((size_t)t * NN + node) * 2 + 1];
  v16h aGemb;
#pragma unroll
  for (int e = 0; e < 16; ++e) {
    int k = kbase + e + (e >= 8 ? 8 : 0);
    float v = f0 * sWg[2 * k] + f1 * sWg[2 * k + 1] + sBg[k];
    aGemb[e] = (h16)(v > 0.0f ? v : 0.0f);
  }

  const h16* ghrow = gh16 + (size_t)node * GH;  // old gh (f16 shadow)
  v16h aH0 = loadA16(ghrow, kbase, 0);
  v16h aH1 = loadA16(ghrow, kbase, 32);

  // gates[16 x 256] = gemb@Wih^T + gh@Whh^T + (bih+bhh)
  v8f acc[16];
#pragma unroll
  for (int ct = 0; ct < 16; ++ct) {
    float cb = bg[ct * 16 + mrow];
    v8f c = {cb, cb, cb, cb, cb, cb, cb, cb};
    c = wmma16(aGemb, loadB16(pkIHg, ct, lane), c);
    c = wmma16(aH0, loadB16(pkHHg, ct, lane), c);
    c = wmma16(aH1, loadB16(pkHHg, 16 + ct, lane), c);
    acc[ct] = c;
  }

  // elementwise LSTM update, fully register-local (i=acc[0..3] f=4..7 g=8..11 o=12..15)
  float s0 = 0.f, s1 = 0.f, s2 = 0.f, s3 = 0.f;
#pragma unroll
  for (int j = 0; j < 4; ++j) {
#pragma unroll
    for (int r = 0; r < 8; ++r) {
      int nrow = base + r + 8 * hs;
      int chc = 16 * j + mrow;
      size_t idx = (size_t)nrow * GH + chc;
      float cold = gcf[idx];
      float iv = sigmoidf_(acc[j][r]);
      float fv = sigmoidf_(acc[4 + j][r]);
      float gv = tanhf(acc[8 + j][r]);
      float ov = sigmoidf_(acc[12 + j][r]);
      float c2 = fv * cold + iv * gv;
      float h2 = ov * tanhf(c2);
      gcf[idx] = c2;
      ghf[idx] = h2;
      gh16[idx] = (h16)h2;           // old values already consumed above
      if (j == 0) s0 += h2; else if (j == 1) s1 += h2;
      else if (j == 2) s2 += h2; else s3 += h2;
    }
  }
  // wave-local sum over rows -> per-channel; combine lane halves; stage to LDS
  s0 += __shfl_xor(s0, 16, 32);
  s1 += __shfl_xor(s1, 16, 32);
  s2 += __shfl_xor(s2, 16, 32);
  s3 += __shfl_xor(s3, 16, 32);
  if (hs == 0) {
    sSum[widx * GH + 0 * 16 + mrow] = s0;
    sSum[widx * GH + 1 * 16 + mrow] = s1;
    sSum[widx * GH + 2 * 16 + mrow] = s2;
    sSum[widx * GH + 3 * 16 + mrow] = s3;
  }
  __syncthreads();
  if (tid < GH) {
    float p = 0.0f;
#pragma unroll
    for (int w = 0; w < WPB; ++w) p += sSum[w * GH + tid];   // fixed order
    partials[(size_t)blockIdx.x * GH + tid] = p;
  }
}

// ---- per-step kernel 2: deterministic global reduction + sum@w_graph -------
__global__ void reduce_gh(const float* __restrict__ partials, int nblk,
                          const float* __restrict__ wgraph, float* __restrict__ sumW) {
  __shared__ float sG[GH];
  int tid = threadIdx.x;  // 64 threads
  float s = 0.0f;
  for (int b = 0; b < nblk; ++b) s += partials[(size_t)b * GH + tid];  // fixed order
  sG[tid] = s;
  __syncthreads();
  if (tid < 32) {
    float a = 0.0f;
    for (int k = 0; k < GH; ++k) a += sG[k] * wgraph[k * 32 + tid];
    sumW[tid] = a;
  }
}

// ---- per-step kernel 3: graph_feat, cell LSTM, output head -----------------
__global__ void __launch_bounds__(256) cell_step(
    const float* __restrict__ in_data, int t,
    const float* __restrict__ Wdyn, const float* __restrict__ bdyn,
    const h16* __restrict__ gh16, const float* __restrict__ sumW,
    h16* ch16, float* chf, float* ccf,
    const h16* __restrict__ pkWg, const h16* __restrict__ pkIHc,
    const h16* __restrict__ pkHHc, const h16* __restrict__ pkOut,
    const float* __restrict__ bc, const float* __restrict__ bout,
    float* __restrict__ out) {
  __shared__ float sWd[64], sBd[32];
  __shared__ __align__(16) h16 gfS[WPB * 16 * 32];  // graph_feat f16 staging
  __shared__ __align__(16) h16 chS[WPB * 16 * 64];  // new ch f16 staging

  int tid = threadIdx.x, lane = tid & 31, widx = tid >> 5;
  if (tid < 64) sWd[tid] = Wdyn[tid];
  if (tid < 32) sBd[tid] = bdyn[tid];
  __syncthreads();

  int base = (blockIdx.x * WPB + widx) * 16;
  int mrow = lane & 15, hs = lane >> 4, kbase = hs * 8;
  int node = base + mrow;

  // ---- graph_feat = sumW - gh @ w_graph  (gh is NEW gh from graph_step) ----
  const h16* ghrow = gh16 + (size_t)node * GH;
  v16h aH0 = loadA16(ghrow, kbase, 0);
  v16h aH1 = loadA16(ghrow, kbase, 32);
  v8f gf0 = {0.f, 0.f, 0.f, 0.f, 0.f, 0.f, 0.f, 0.f}, gf1 = gf0;
  gf0 = wmma16(aH0, loadB16(pkWg, 0, lane), gf0);
  gf0 = wmma16(aH1, loadB16(pkWg, 2, lane), gf0);
  gf1 = wmma16(aH0, loadB16(pkWg, 1, lane), gf1);
  gf1 = wmma16(aH1, loadB16(pkWg, 3, lane), gf1);
  // D layout -> LDS (f16) so it can be re-read in A layout
#pragma unroll
  for (int r = 0; r < 8; ++r) {
    int mm = r + 8 * hs;
    int n0 = 0 * 16 + mrow, n1 = 1 * 16 + mrow;
    gfS[widx * 512 + mm * 32 + n0] = (h16)(sumW[n0] - gf0[r]);
    gfS[widx * 512 + mm * 32 + n1] = (h16)(sumW[n1] - gf1[r]);
  }
  __syncthreads();

  // ---- cell_emb A operands: K 0..31 = dyn (in-lane), K 32..63 = graph_feat -
  float f0 = in_data[((size_t)t * NN + node) * 2 + 0];
  float f1 = in_data[((size_t)t * NN + node) * 2 + 1];
  v16h aCE0;
#pragma unroll
  for (int e = 0; e < 16; ++e) {
    int k = kbase + e + (e >= 8 ? 8 : 0);
    float v = f0 * sWd[2 * k] + f1 * sWd[2 * k + 1] + sBd[k];
    aCE0[e] = (h16)(v > 0.0f ? v : 0.0f);
  }
  v16h aCE1 = loadA16(&gfS[widx * 512 + mrow * 32], kbase, 0);

  const h16* chrow = ch16 + (size_t)node * CH;  // old ch
  v16h aC0 = loadA16(chrow, kbase, 0);
  v16h aC1 = loadA16(chrow, kbase, 32);

  // ---- gates[16 x 256] = ce@Wih_c^T + ch@Whh_c^T + (bih+bhh) ---------------
  v8f acc[16];
#pragma unroll
  for (int ct = 0; ct < 16; ++ct) {
    float cb = bc[ct * 16 + mrow];
    v8f c = {cb, cb, cb, cb, cb, cb, cb, cb};
    c = wmma16(aCE0, loadB16(pkIHc, ct, lane), c);
    c = wmma16(aCE1, loadB16(pkIHc, 16 + ct, lane), c);
    c = wmma16(aC0, loadB16(pkHHc, ct, lane), c);
    c = wmma16(aC1, loadB16(pkHHc, 16 + ct, lane), c);
    acc[ct] = c;
  }

  // ---- elementwise LSTM update --------------------------------------------
#pragma unroll
  for (int j = 0; j < 4; ++j) {
#pragma unroll
    for (int r = 0; r < 8; ++r) {
      int mm = r + 8 * hs;
      int nrow = base + mm;
      int chc = 16 * j + mrow;
      size_t idx = (size_t)nrow * CH + chc;
      float cold = ccf[idx];
      float iv = sigmoidf_(acc[j][r]);
      float fv = sigmoidf_(acc[4 + j][r]);
      float gv = tanhf(acc[8 + j][r]);
      float ov = sigmoidf_(acc[12 + j][r]);
      float c2 = fv * cold + iv * gv;
      float h2 = ov * tanhf(c2);
      ccf[idx] = c2;
      chf[idx] = h2;
      ch16[idx] = (h16)h2;
      chS[widx * 1024 + mm * 64 + chc] = (h16)h2;
    }
  }
  __syncthreads();

  // ---- out = ch_new @ W_out^T + b_out  (padded 16-col WMMA) ----------------
  v16h aN0 = loadA16(&chS[widx * 1024 + mrow * 64], kbase, 0);
  v16h aN1 = loadA16(&chS[widx * 1024 + mrow * 64], kbase, 32);
  float ob = (mrow < OUTW) ? bout[mrow] : 0.0f;
  v8f oc = {ob, ob, ob, ob, ob, ob, ob, ob};
  oc = wmma16(aN0, loadB16(pkOut, 0, lane), oc);
  oc = wmma16(aN1, loadB16(pkOut, 1, lane), oc);
#pragma unroll
  for (int r = 0; r < 8; ++r) {
    int mm = r + 8 * hs;
    if (mrow < OUTW)
      out[((size_t)t * NN + base + mm) * OUTW + mrow] = oc[r];
  }
}

// ---- final state copy into d_out tail --------------------------------------
__global__ void finalize_states(const float* __restrict__ chf, const float* __restrict__ ccf,
                                const float* __restrict__ ghf, const float* __restrict__ gcf,
                                float* __restrict__ out, int n) {
  int i = blockIdx.x * blockDim.x + threadIdx.x;
  if (i < n) {
    float* p = out + (size_t)SEQ * NN * OUTW;
    p[i] = chf[i];
    p[(size_t)n + i] = ccf[i];
    p[(size_t)2 * n + i] = ghf[i];
    p[(size_t)3 * n + i] = gcf[i];
  }
}

// ---------------------------------------------------------------------------
extern "C" void kernel_launch(void* const* d_in, const int* in_sizes, int n_in,
                              void* d_out, int out_size, void* d_ws, size_t ws_size,
                              hipStream_t stream) {
  const float* in_data = (const float*)d_in[0];
  const float* ch0   = (const float*)d_in[1];
  const float* cc0   = (const float*)d_in[2];
  const float* gh0   = (const float*)d_in[3];
  const float* gc0   = (const float*)d_in[4];
  const float* Wdyn  = (const float*)d_in[5];
  const float* bdyn  = (const float*)d_in[6];
  const float* Wgemb = (const float*)d_in[7];
  const float* bgemb = (const float*)d_in[8];
  const float* WihG  = (const float*)d_in[9];
  const float* WhhG  = (const float*)d_in[10];
  const float* bihG  = (const float*)d_in[11];
  const float* bhhG  = (const float*)d_in[12];
  const float* wgr   = (const float*)d_in[13];
  const float* WihC  = (const float*)d_in[14];
  const float* WhhC  = (const float*)d_in[15];
  const float* bihC  = (const float*)d_in[16];
  const float* bhhC  = (const float*)d_in[17];
  const float* Wout  = (const float*)d_in[18];
  const float* bout  = (const float*)d_in[19];
  float* out = (float*)d_out;

  // workspace carve-up (~80.3 MB total)
  char* ws = (char*)d_ws;
  size_t off = 0;
  auto take = [&](size_t bytes) { char* p = ws + off; off += (bytes + 255) & ~(size_t)255; return p; };
  float* ghf = (float*)take((size_t)NN * GH * 4);
  float* gcf = (float*)take((size_t)NN * GH * 4);
  float* chf = (float*)take((size_t)NN * CH * 4);
  float* ccf = (float*)take((size_t)NN * CH * 4);
  h16* gh16 = (h16*)take((size_t)NN * GH * 2);
  h16* ch16 = (h16*)take((size_t)NN * CH * 2);
  float* partials = (float*)take((size_t)NBLK * GH * 4);
  float* sumW = (float*)take(32 * 4);
  h16* pkIHg = (h16*)take(1 * 16 * 512 * 2);
  h16* pkHHg = (h16*)take(2 * 16 * 512 * 2);
  h16* pkIHc = (h16*)take(2 * 16 * 512 * 2);
  h16* pkHHc = (h16*)take(2 * 16 * 512 * 2);
  h16* pkWg  = (h16*)take(2 * 2 * 512 * 2);
  h16* pkOut = (h16*)take(2 * 1 * 512 * 2);
  float* bg = (float*)take(256 * 4);
  float* bc = (float*)take(256 * 4);
  (void)in_sizes; (void)n_in; (void)out_size; (void)ws_size;

  int nState = NN * GH;
  init_states<<<(nState + 255) / 256, 256, 0, stream>>>(
      ch0, cc0, gh0, gc0, ghf, gcf, chf, ccf, gh16, ch16, nState);
  pack_weights<<<1, 256, 0, stream>>>(
      WihG, WhhG, bihG, bhhG, wgr, WihC, WhhC, bihC, bhhC, Wout,
      pkIHg, pkHHg, pkIHc, pkHHc, pkWg, pkOut, bg, bc);

  for (int t = 0; t < SEQ; ++t) {
    graph_step<<<NBLK, 256, 0, stream>>>(
        in_data, t, Wgemb, bgemb, gh16, ghf, gcf, pkIHg, pkHHg, bg, partials);
    reduce_gh<<<1, 64, 0, stream>>>(partials, NBLK, wgr, sumW);
    cell_step<<<NBLK, 256, 0, stream>>>(
        in_data, t, Wdyn, bdyn, gh16, sumW, ch16, chf, ccf,
        pkWg, pkIHc, pkHHc, pkOut, bc, bout, out);
  }
  finalize_states<<<(nState + 255) / 256, 256, 0, stream>>>(
      chf, ccf, ghf, gcf, out, nState);
}
